// MetaVisualLearner_44023414784293
// MI455X (gfx1250) — compile-verified, matrix operand order checked by reading, not run on hardware
//
#include <hip/hip_runtime.h>
#include <hip/hip_bf16.h>
#include <math.h>

typedef __attribute__((ext_vector_type(16))) __bf16 v16bf;
typedef __attribute__((ext_vector_type(2)))  __bf16 v2bf;
typedef __attribute__((ext_vector_type(8)))  float  v8f;

#define BB     2
#define NN     16384
#define KKE    32
#define DD     128
#define KEYD   64
#define NEDGES (BB*NN*KKE)          /* 1048576 */

/* ws layout in __bf16 elements: fragment-major swizzled weights (A = W^T) */
#define ENC1_OFF  0                  /* 256->128 : 8 ot x 8 kt frags */
#define ENCH_OFF  32768              /* 3 x (128->128): 8x4 frags    */
#define ENCO_OFF  81920              /* 128->64 : 4x4 frags          */
#define BIAS1_OFF 90112              /* 256->128                      */
#define BIASH_OFF 122880             /* 3 x (128->128)                */

__device__ __forceinline__ __bf16 f2bf(float f) {
    unsigned u = __builtin_bit_cast(unsigned, f);
    unsigned r = (u + 0x7fffu + ((u >> 16) & 1u)) >> 16;   // RNE
    unsigned short s = (unsigned short)r;
    return __builtin_bit_cast(__bf16, s);
}

/* hardware packed f32->bf16 (RNE): one VALU op per two elements */
__device__ __forceinline__ v2bf pk_bf(float a, float b) {
    v2bf r;
    asm("v_cvt_pk_bf16_f32 %0, %1, %2" : "=v"(r) : "v"(a), "v"(b));
    return r;
}

#if __has_builtin(__builtin_amdgcn_rcpf)
__device__ __forceinline__ float fast_rcp(float x) { return __builtin_amdgcn_rcpf(x); }
#else
__device__ __forceinline__ float fast_rcp(float x) { return 1.0f / x; }
#endif

__device__ __forceinline__ float sigmoid_fast(float y) {
    return fast_rcp(1.0f + __expf(-y));          // v_exp_f32 + v_rcp_f32
}

/* gelu(x) = 0.5x(1+tanh(c(x+0.044715x^3))) = x * sigmoid(2c(x+0.044715x^3)) */
__device__ __forceinline__ float gelu_fast(float x) {
    float x2 = x * x;
    float z2 = x * (1.5957691216057308f + 0.07135481627262f * x2);
    return x * sigmoid_fast(z2);
}

union FragU { v16bf v; v2bf h[8]; };

__device__ __forceinline__ v16bf load_frag(const __bf16* p, int lane) {
    return *(const v16bf*)(p + lane * 16);   // 32B per lane, contiguous
}

__device__ __forceinline__ v8f wmma_bf16(v16bf a, v16bf b, v8f c) {
    return __builtin_amdgcn_wmma_f32_16x16x32_bf16(
        false, a, false, b, (short)0, c, false, false);
}

/* One MLP layer as WMMA stream with PIPE-deep A-fragment software pipeline.
   Sequence s = kt*NOT + ot ; ws fragment offset = (ot*NKT + kt)*512. */
template <int NKT, int NOT, int PIPE>
__device__ __forceinline__ void mlp_layer(const __bf16* wsbase, int lane,
                                          const v16bf* B, v8f* acc) {
    constexpr int NF = NKT * NOT;
#pragma unroll
    for (int t = 0; t < NOT; ++t) acc[t] = (v8f){0,0,0,0,0,0,0,0};
    v16bf pipe[PIPE];
#pragma unroll
    for (int i = 0; i < PIPE; ++i)
        pipe[i] = load_frag(wsbase + ((i % NOT) * NKT + (i / NOT)) * 512, lane);
#pragma unroll
    for (int s = 0; s < NF; ++s) {
        v16bf cur = pipe[s % PIPE];
        const int sn = s + PIPE;
        if (sn < NF)
            pipe[s % PIPE] =
                load_frag(wsbase + ((sn % NOT) * NKT + (sn / NOT)) * 512, lane);
        acc[s % NOT] = wmma_bf16(cur, B[s / NOT], acc[s % NOT]);
    }
}

/* Build layer-1 B fragments: lane<16 holds edge (lane) dims {32kt+0..7, 32kt+16..23},
   lane>=16 holds dims {+8..15, +24..31}. kt 0..3 -> x row, 4..7 -> y row. */
__device__ __forceinline__ void build_in_frags(const float* fx, const float* fy,
                                               int off, v16bf (&inB)[8]) {
#pragma unroll
    for (int kt = 0; kt < 8; ++kt) {
        const float* src = (kt < 4) ? fx : fy;
        int d0 = ((kt & 3) << 5) + off;
        float4 a0 = *(const float4*)(src + d0);
        float4 a1 = *(const float4*)(src + d0 + 4);
        float4 c0 = *(const float4*)(src + d0 + 16);
        float4 c1 = *(const float4*)(src + d0 + 20);
        FragU fu;
        fu.h[0] = pk_bf(a0.x, a0.y); fu.h[1] = pk_bf(a0.z, a0.w);
        fu.h[2] = pk_bf(a1.x, a1.y); fu.h[3] = pk_bf(a1.z, a1.w);
        fu.h[4] = pk_bf(c0.x, c0.y); fu.h[5] = pk_bf(c0.z, c0.w);
        fu.h[6] = pk_bf(c1.x, c1.y); fu.h[7] = pk_bf(c1.z, c1.w);
        inB[kt] = fu.v;
    }
}

/* bias-add + activation on 8 output tiles (128 dims), repack as 4 B-fragments */
template <bool GELU>
__device__ __forceinline__ void act_refrag(const v8f (&a)[8], const float* bias,
                                           int off, v16bf (&Bf)[4]) {
    float h[64];
#pragma unroll
    for (int t = 0; t < 8; ++t) {
        float4 b0 = *(const float4*)(bias + 16 * t + off);
        float4 b1 = *(const float4*)(bias + 16 * t + off + 4);
        float bs[8] = {b0.x, b0.y, b0.z, b0.w, b1.x, b1.y, b1.z, b1.w};
#pragma unroll
        for (int r = 0; r < 8; ++r) {
            float v = a[t][r] + bs[r];
            h[t * 8 + r] = GELU ? gelu_fast(v) : fmaxf(v, 0.0f);
        }
    }
#pragma unroll
    for (int kt = 0; kt < 4; ++kt) {
        FragU fu;
#pragma unroll
        for (int j = 0; j < 4; ++j) {
            fu.h[j]     = pk_bf(h[(2 * kt) * 8 + 2 * j],     h[(2 * kt) * 8 + 2 * j + 1]);
            fu.h[4 + j] = pk_bf(h[(2 * kt + 1) * 8 + 2 * j], h[(2 * kt + 1) * 8 + 2 * j + 1]);
        }
        Bf[kt] = fu.v;
    }
}

__global__ void __launch_bounds__(256)
prep_weights(const float* __restrict__ src, __bf16* __restrict__ dst,
             int IN, int OUT) {
    int i = blockIdx.x * blockDim.x + threadIdx.x;
    if (i >= IN * OUT) return;
    int fi = i >> 9, lp = i & 511;
    int lane = lp >> 4, p = lp & 15;
    int nkt = IN >> 5;
    int ot = fi / nkt, kt = fi - ot * nkt;
    int o = (ot << 4) + (lane & 15);
    int ioff = (p < 8 ? p : p + 8) + ((lane >= 16) ? 8 : 0);
    int iin = (kt << 5) + ioff;
    dst[i] = f2bf(src[iin * OUT + o]);   // A = W^T : element (out=o, in=iin)
}

__global__ void __launch_bounds__(256)
mvl_main(const float* __restrict__ bbf, const float* __restrict__ ga,
         const float* __restrict__ embed,
         const float* __restrict__ enc_b_in,  const float* __restrict__ enc_b_hid,
         const float* __restrict__ enc_b_out,
         const float* __restrict__ bias_b_in, const float* __restrict__ bias_b_hid,
         const float* __restrict__ bias_w_out, const float* __restrict__ bias_b_out,
         const int* __restrict__ idx, const __bf16* __restrict__ ws,
         float* __restrict__ out) {
    const int lane  = threadIdx.x & 31;
    const int wave  = threadIdx.x >> 5;
    const int tile  = blockIdx.x * 8 + wave;
    const int e     = tile * 16 + (lane & 15);
    const bool upper = lane >= 16;
    const int off   = upper ? 8 : 0;

    const int b  = e >> 19;                           // N*K = 2^19
    const int xi = idx[NEDGES + e];
    const int yi = idx[2 * NEDGES + e];
    const float* fx = bbf + (size_t)(b * NN + xi) * DD;
    const float* fy = bbf + (size_t)(b * NN + yi) * DD;

    /* ---------------- encoder MLP (relu) ---------------- */
    v8f acc[8];
    {
        v16bf inB[8];
        build_in_frags(fx, fy, off, inB);
        mlp_layer<8, 8, 4>(ws + ENC1_OFF, lane, inB, acc);
    }
    v16bf hB[4];
    act_refrag<false>(acc, enc_b_in, off, hB);
#pragma unroll
    for (int L = 0; L < 3; ++L) {
        mlp_layer<4, 8, 4>(ws + ENCH_OFF + L * 16384, lane, hB, acc);
        act_refrag<false>(acc, enc_b_hid + L * 128, off, hB);
    }
    v8f ao[4];
    mlp_layer<4, 4, 4>(ws + ENCO_OFF, lane, hB, ao);

    float cond[32];
    float ssq = 0.0f;
#pragma unroll
    for (int t = 0; t < 4; ++t) {
        float4 b0 = *(const float4*)(enc_b_out + 16 * t + off);
        float4 b1 = *(const float4*)(enc_b_out + 16 * t + off + 4);
        float bs[8] = {b0.x, b0.y, b0.z, b0.w, b1.x, b1.y, b1.z, b1.w};
#pragma unroll
        for (int r = 0; r < 8; ++r) {
            float v = ao[t][r] + bs[r];
            cond[t * 8 + r] = v;
            ssq += v * v;
        }
    }
    ssq += __shfl_xor(ssq, 16, 32);
    float inv_c = fast_rcp(fmaxf(sqrtf(ssq), 1e-12f));

    float sig[2];
#pragma unroll
    for (int m = 0; m < 2; ++m) {
        float es = 0.0f, dt = 0.0f;
#pragma unroll
        for (int t = 0; t < 4; ++t) {
            float4 e0 = *(const float4*)(embed + m * KEYD + 16 * t + off);
            float4 e1 = *(const float4*)(embed + m * KEYD + 16 * t + off + 4);
            float ev[8] = {e0.x, e0.y, e0.z, e0.w, e1.x, e1.y, e1.z, e1.w};
#pragma unroll
            for (int r = 0; r < 8; ++r) {
                es += ev[r] * ev[r];
                dt += cond[t * 8 + r] * ev[r];
            }
        }
        es += __shfl_xor(es, 16, 32);
        dt += __shfl_xor(dt, 16, 32);
        float z = dt * inv_c * fast_rcp(fmaxf(sqrtf(es), 1e-12f));
        sig[m] = sigmoid_fast(z);
    }

    /* ---------------- bias MLP (gelu) ---------------- */
    {
        v16bf inB[8];
        build_in_frags(fx, fy, off, inB);
        mlp_layer<8, 8, 4>(ws + BIAS1_OFF, lane, inB, acc);
    }
    act_refrag<true>(acc, bias_b_in, off, hB);
#pragma unroll
    for (int L = 0; L < 3; ++L) {
        mlp_layer<4, 8, 4>(ws + BIASH_OFF + L * 16384, lane, hB, acc);
        if (L < 2) act_refrag<true>(acc, bias_b_hid + L * 128, off, hB);
    }
    /* last hidden: gelu then dot with w_out (128-dim split across lane pair) */
    float db = 0.0f;
#pragma unroll
    for (int t = 0; t < 8; ++t) {
        float4 b0 = *(const float4*)(bias_b_hid + 2 * 128 + 16 * t + off);
        float4 b1 = *(const float4*)(bias_b_hid + 2 * 128 + 16 * t + off + 4);
        float4 w0 = *(const float4*)(bias_w_out + 16 * t + off);
        float4 w1 = *(const float4*)(bias_w_out + 16 * t + off + 4);
        float bs[8] = {b0.x, b0.y, b0.z, b0.w, b1.x, b1.y, b1.z, b1.w};
        float wv[8] = {w0.x, w0.y, w0.z, w0.w, w1.x, w1.y, w1.z, w1.w};
#pragma unroll
        for (int r = 0; r < 8; ++r)
            db += gelu_fast(acc[t][r] + bs[r]) * wv[r];
    }
    db += __shfl_xor(db, 16, 32);
    float eb = db + bias_b_out[0];

    /* ---------------- combine ---------------- */
    int rem = e & (NN * KKE - 1);
    float res = 0.0f;
#pragma unroll
    for (int m = 0; m < 2; ++m) {
        float g = ga[(size_t)(b * 2 + m) * (NN * KKE) + rem];
        res += sig[m] * (g - eb);
    }
    if (!upper) out[e] = res;
}

extern "C" void kernel_launch(void* const* d_in, const int* in_sizes, int n_in,
                              void* d_out, int out_size, void* d_ws, size_t ws_size,
                              hipStream_t stream) {
    const float* bbf        = (const float*)d_in[0];
    const float* ga         = (const float*)d_in[1];
    const float* embed      = (const float*)d_in[2];
    const float* enc_w_in   = (const float*)d_in[3];
    const float* enc_b_in   = (const float*)d_in[4];
    const float* enc_w_hid  = (const float*)d_in[5];
    const float* enc_b_hid  = (const float*)d_in[6];
    const float* enc_w_out  = (const float*)d_in[7];
    const float* enc_b_out  = (const float*)d_in[8];
    const float* bias_w_in  = (const float*)d_in[9];
    const float* bias_b_in  = (const float*)d_in[10];
    const float* bias_w_hid = (const float*)d_in[11];
    const float* bias_b_hid = (const float*)d_in[12];
    const float* bias_w_out = (const float*)d_in[13];
    const float* bias_b_out = (const float*)d_in[14];
    const int*   idx        = (const int*)d_in[15];
    __bf16* ws  = (__bf16*)d_ws;
    float*  out = (float*)d_out;

    auto prep = [&](const float* src, int dstoff, int IN, int OUT) {
        int total = IN * OUT;
        prep_weights<<<(total + 255) / 256, 256, 0, stream>>>(src, ws + dstoff, IN, OUT);
    };
    prep(enc_w_in,  ENC1_OFF, 256, 128);
    for (int i = 0; i < 3; ++i)
        prep(enc_w_hid + i * 128 * 128, ENCH_OFF + i * 16384, 128, 128);
    prep(enc_w_out, ENCO_OFF, 128, 64);
    prep(bias_w_in, BIAS1_OFF, 256, 128);
    for (int i = 0; i < 3; ++i)
        prep(bias_w_hid + i * 128 * 128, BIASH_OFF + i * 16384, 128, 128);

    const int tiles  = NEDGES / 16;          // 65536 wave-tiles
    const int blocks = tiles / 8;            // 8 waves per block
    mvl_main<<<blocks, 256, 0, stream>>>(bbf, ga, embed,
                                         enc_b_in, enc_b_hid, enc_b_out,
                                         bias_b_in, bias_b_hid, bias_w_out, bias_b_out,
                                         idx, ws, out);
}
